// Butterfly_83494164234553
// MI455X (gfx1250) — compile-verified, matrix operand order checked by reading, not run on hardware
//
#include <hip/hip_runtime.h>

// ---------------------------------------------------------------------------
// Butterfly network (N=1024, B=16384, DEPTH=2), recast as:
//   1) build dense W (1024x1024) by pushing identity rows through the factor
//      chain (linear operator; imaginary input is zero so out = x @ Re(W) + b)
//   2) out = x @ W + b as a bf16 hi/lo split GEMM on v_wmma_f32_16x16x32_bf16,
//      double-buffered LDS, async W staging, register-prefetched x staging.
// ---------------------------------------------------------------------------

#define NN        1024
#define BATCH     16384
#define TOTAL_AB  4092   // sum_{m=2..1024} 2m
#define BM        128
#define BN        64
#define BK        32
#define KCHUNKS   (NN / BK)

typedef __attribute__((ext_vector_type(16))) __bf16 v16bf;
typedef __attribute__((ext_vector_type(8)))  __bf16 v8bf;
typedef __attribute__((ext_vector_type(8)))  float  v8f;
typedef __attribute__((ext_vector_type(4)))  int    v4i;

#if defined(__AMDGCN__) && \
    __has_builtin(__builtin_amdgcn_global_load_async_to_lds_b128) && \
    __has_builtin(__builtin_amdgcn_s_wait_asynccnt)
#define USE_ASYNC 1
// ROCm 7.2 clang-22 signature (from diagnostics): (v4i* src, v4i* dst, Ii, Ii)
#define ASYNC_B128(src, dst) \
    __builtin_amdgcn_global_load_async_to_lds_b128((v4i*)(src), (v4i*)(dst), 0, 0)
#else
#define USE_ASYNC 0
#endif

// ---------------------------------------------------------------------------
// Kernel 1: build W^T split into bf16 hi/lo planes.
// One workgroup per basis row j: h = e_j -> factor chain -> W[j,:], stored
// transposed: wt[n][j] so the GEMM's B-fragments read contiguous K.
// ---------------------------------------------------------------------------
__global__ __launch_bounds__(256) void build_wt_kernel(
    const float* __restrict__ perm_logit,  // (2,3)
    const float* __restrict__ abcd,        // (2, 4092, 2)
    __bf16* __restrict__ wt_hi,            // (N, N): [n][j]
    __bf16* __restrict__ wt_lo)
{
    __shared__ float2 buf0[NN];
    __shared__ float2 buf1[NN];
    const int row = blockIdx.x;
    const int tid = threadIdx.x;
    float2* cur = buf0;
    float2* nxt = buf1;

    for (int e = tid; e < NN; e += 256)
        cur[e] = make_float2(e == row ? 1.0f : 0.0f, 0.0f);
    __syncthreads();

    for (int d = 0; d < 2; ++d) {
        const float p0 = 1.0f / (1.0f + expf(-perm_logit[d * 3 + 0]));
        const float p1 = 1.0f / (1.0f + expf(-perm_logit[d * 3 + 1]));
        const float p2 = 1.0f / (1.0f + expf(-perm_logit[d * 3 + 2]));

        // ---- perm factors, m = 4 .. 1024 (reversed(perm_sizes)) ----
        for (int m = 4; m <= NN; m <<= 1) {
            const int half = m >> 1;
            // pass 1: even/odd separation mix
            for (int g = tid; g < NN; g += 256) {
                const int j    = g & (m - 1);
                const int base = g - j;
                const int src  = (j < half) ? (2 * j) : (2 * (j - half) + 1);
                const float2 a = cur[g];
                const float2 s = cur[base + src];
                nxt[g] = make_float2((1.0f - p0) * a.x + p0 * s.x,
                                     (1.0f - p0) * a.y + p0 * s.y);
            }
            __syncthreads();
            { float2* t = cur; cur = nxt; nxt = t; }
            // pass 2: half-reversal mixes (p1 first half, p2 second half)
            for (int g = tid; g < NN; g += 256) {
                const int j    = g & (m - 1);
                const int base = g - j;
                int partner; float p;
                if (j < half) { partner = half - 1 - j;     p = p1; }
                else          { partner = 3 * half - 1 - j; p = p2; }
                const float2 a = cur[g];
                const float2 c = cur[base + partner];
                nxt[g] = make_float2((1.0f - p) * a.x + p * c.x,
                                     (1.0f - p) * a.y + p * c.y);
            }
            __syncthreads();
            { float2* t = cur; cur = nxt; nxt = t; }
        }

        // ---- diag factors, m = 2 .. 1024 (reversed(offs)) ----
        for (int m = 2; m <= NN; m <<= 1) {
            const int half = m >> 1;
            const int off  = 4096 - 4 * m;  // offset of factor m in abcd
            const float* ab = abcd + ((size_t)d * TOTAL_AB + off) * 2;
            for (int g = tid; g < NN; g += 256) {
                const int j    = g & (m - 1);
                const int base = g - j;
                const int i    = (j >= half) ? 1 : 0;
                const int k    = j & (half - 1);
                const float2 u = cur[base + k];
                const float2 v = cur[base + half + k];
                const float* A  = ab + (((size_t)(i * 2 + 0) * half + k) * 2);
                const float* Bc = ab + (((size_t)(i * 2 + 1) * half + k) * 2);
                const float Ar = A[0],  Ai = A[1];
                const float Br = Bc[0], Bi = Bc[1];
                nxt[g] = make_float2(Ar * u.x - Ai * u.y + Br * v.x - Bi * v.y,
                                     Ar * u.y + Ai * u.x + Br * v.y + Bi * v.x);
            }
            __syncthreads();
            { float2* t = cur; cur = nxt; nxt = t; }
        }
    }

    // Store transposed, split into bf16 hi + lo (residual) planes.
    for (int n = tid; n < NN; n += 256) {
        const float v  = cur[n].x;          // real part only (input imag == 0)
        const __bf16 h = (__bf16)v;
        const __bf16 l = (__bf16)(v - (float)h);
        wt_hi[(size_t)n * NN + row] = h;
        wt_lo[(size_t)n * NN + row] = l;
    }
}

// ---------------------------------------------------------------------------
// Kernel 2: out[B,N] = x[B,N] @ W[N,N] + b, bf16 hi/lo split WMMA GEMM.
// Block tile 128x64, 8 waves (4x2), 32x32 per wave (2x2 of 16x16 wmma tiles).
// Ping-pong LDS; W tiles staged by async-to-LDS (ASYNCcnt), x tiles
// register-prefetched and split fp32 -> bf16 hi/lo on the fly.
// ---------------------------------------------------------------------------
__global__ __launch_bounds__(256) void gemm_wmma_kernel(
    const float*  __restrict__ x,       // (B, N)
    const __bf16* __restrict__ wt_hi,   // (N, N) [n][k]
    const __bf16* __restrict__ wt_lo,
    const float*  __restrict__ bias,    // (N)
    float*        __restrict__ out)     // (B, N)
{
    __shared__ __attribute__((aligned(16))) __bf16 xh[2][BM * BK];
    __shared__ __attribute__((aligned(16))) __bf16 xl[2][BM * BK];
    __shared__ __attribute__((aligned(16))) __bf16 wh[2][BN * BK];
    __shared__ __attribute__((aligned(16))) __bf16 wl[2][BN * BK];

    const int tid   = threadIdx.x;
    const int lane  = tid & 31;
    const int wave  = tid >> 5;
    const int waveM = wave >> 1;   // 0..3
    const int waveN = wave & 1;    // 0..1
    const int rowBase = blockIdx.x * BM;
    const int colBase = blockIdx.y * BN;

    v8f acc[2][2] = {{{0.f,0.f,0.f,0.f,0.f,0.f,0.f,0.f},
                      {0.f,0.f,0.f,0.f,0.f,0.f,0.f,0.f}},
                     {{0.f,0.f,0.f,0.f,0.f,0.f,0.f,0.f},
                      {0.f,0.f,0.f,0.f,0.f,0.f,0.f,0.f}}};

    // Cooperative-load coordinates.
    const int xr = tid >> 1;            // 0..127, x-tile row
    const int xc = (tid & 1) * 16;      // 0 / 16, 16 consecutive floats
    const int wr = tid >> 2;            // 0..63, w-tile row (n)
    const int wc = (tid & 3) * 8;       // 8 consecutive bf16 (16 bytes)

    const float*  xg  = x     + (size_t)(rowBase + xr) * NN + xc;
    const __bf16* wgh = wt_hi + (size_t)(colBase + wr) * NN + wc;
    const __bf16* wgl = wt_lo + (size_t)(colBase + wr) * NN + wc;
    const int wd = wr * BK + wc;        // W dest offset in LDS tile

    float xv[16];
#if !USE_ASYNC
    v8bf wvh, wvl;
#endif

    // ---------------- prologue: stage chunk 0 into buffer 0 ----------------
#if USE_ASYNC
    ASYNC_B128(wgh, &wh[0][wd]);
    ASYNC_B128(wgl, &wl[0][wd]);
#else
    wvh = *(const v8bf*)wgh;
    wvl = *(const v8bf*)wgl;
#endif
    #pragma unroll
    for (int i = 0; i < 16; ++i) xv[i] = xg[i];

    #pragma unroll
    for (int i = 0; i < 16; ++i) {
        const __bf16 h = (__bf16)xv[i];
        xh[0][xr * BK + xc + i] = h;
        xl[0][xr * BK + xc + i] = (__bf16)(xv[i] - (float)h);
    }
#if USE_ASYNC
    __builtin_amdgcn_s_wait_asynccnt(0);
#else
    *(v8bf*)&wh[0][wd] = wvh;
    *(v8bf*)&wl[0][wd] = wvl;
#endif
    __syncthreads();

    // ---------------- main pipelined loop ----------------
    for (int k = 0; k < KCHUNKS; ++k) {
        const int cur  = k & 1;
        const int nxt  = cur ^ 1;
        const bool more = (k + 1) < KCHUNKS;

        // Prefetch chunk k+1 (issues before the WMMA block -> overlapped).
        if (more) {
            const int ko = (k + 1) * BK;
#if USE_ASYNC
            ASYNC_B128(wgh + ko, &wh[nxt][wd]);
            ASYNC_B128(wgl + ko, &wl[nxt][wd]);
#else
            wvh = *(const v8bf*)(wgh + ko);
            wvl = *(const v8bf*)(wgl + ko);
#endif
            #pragma unroll
            for (int i = 0; i < 16; ++i) xv[i] = xg[ko + i];
        }

        // Build fragments per the CDNA5 16-bit wave32 layout:
        // lanes 0-15: K 0-7 then 16-23; lanes 16-31: K 8-15 then 24-31.
        {
            const __bf16* xhc = &xh[cur][0];
            const __bf16* xlc = &xl[cur][0];
            const __bf16* whc = &wh[cur][0];
            const __bf16* wlc = &wl[cur][0];
            const int kh = (lane >> 4) * 8;
            v16bf ahf[2], alf[2], bhf[2], blf[2];
            #pragma unroll
            for (int t = 0; t < 2; ++t) {
                const int mrow = waveM * 32 + t * 16 + (lane & 15);
                const v8bf hA = *(const v8bf*)&xhc[mrow * BK + kh];
                const v8bf hB = *(const v8bf*)&xhc[mrow * BK + kh + 16];
                const v8bf lA = *(const v8bf*)&xlc[mrow * BK + kh];
                const v8bf lB = *(const v8bf*)&xlc[mrow * BK + kh + 16];
                #pragma unroll
                for (int i = 0; i < 8; ++i) {
                    ahf[t][i] = hA[i]; ahf[t][8 + i] = hB[i];
                    alf[t][i] = lA[i]; alf[t][8 + i] = lB[i];
                }
            }
            #pragma unroll
            for (int t = 0; t < 2; ++t) {
                const int ncol = waveN * 32 + t * 16 + (lane & 15);
                const v8bf hA = *(const v8bf*)&whc[ncol * BK + kh];
                const v8bf hB = *(const v8bf*)&whc[ncol * BK + kh + 16];
                const v8bf lA = *(const v8bf*)&wlc[ncol * BK + kh];
                const v8bf lB = *(const v8bf*)&wlc[ncol * BK + kh + 16];
                #pragma unroll
                for (int i = 0; i < 8; ++i) {
                    bhf[t][i] = hA[i]; bhf[t][8 + i] = hB[i];
                    blf[t][i] = lA[i]; blf[t][8 + i] = lB[i];
                }
            }

            // hi*hi + hi*lo + lo*hi  (lo*lo ~ 2^-16 rel, negligible)
            #pragma unroll
            for (int tm = 0; tm < 2; ++tm) {
                #pragma unroll
                for (int tn = 0; tn < 2; ++tn) {
                    acc[tm][tn] = __builtin_amdgcn_wmma_f32_16x16x32_bf16(
                        false, ahf[tm], false, bhf[tn], (short)0, acc[tm][tn], false, false);
                    acc[tm][tn] = __builtin_amdgcn_wmma_f32_16x16x32_bf16(
                        false, ahf[tm], false, blf[tn], (short)0, acc[tm][tn], false, false);
                    acc[tm][tn] = __builtin_amdgcn_wmma_f32_16x16x32_bf16(
                        false, alf[tm], false, bhf[tn], (short)0, acc[tm][tn], false, false);
                }
            }
        }

        // Publish chunk k+1 into the other buffer.
        if (more) {
            #pragma unroll
            for (int i = 0; i < 16; ++i) {
                const __bf16 h = (__bf16)xv[i];
                xh[nxt][xr * BK + xc + i] = h;
                xl[nxt][xr * BK + xc + i] = (__bf16)(xv[i] - (float)h);
            }
#if USE_ASYNC
            __builtin_amdgcn_s_wait_asynccnt(0);
#else
            *(v8bf*)&wh[nxt][wd] = wvh;
            *(v8bf*)&wl[nxt][wd] = wvl;
#endif
            __syncthreads();
        }
    }

    // Epilogue: C/D layout — VGPR r holds M=r (lanes 0-15), M=r+8 (lanes 16-31).
    #pragma unroll
    for (int tm = 0; tm < 2; ++tm) {
        const int rbase = rowBase + waveM * 32 + tm * 16 + ((lane >> 4) * 8);
        #pragma unroll
        for (int tn = 0; tn < 2; ++tn) {
            const int col = colBase + waveN * 32 + tn * 16 + (lane & 15);
            const float bv = bias[col];
            #pragma unroll
            for (int r = 0; r < 8; ++r) {
                out[(size_t)(rbase + r) * NN + col] = acc[tm][tn][r] + bv;
            }
        }
    }
}

// ---------------------------------------------------------------------------
extern "C" void kernel_launch(void* const* d_in, const int* in_sizes, int n_in,
                              void* d_out, int out_size, void* d_ws, size_t ws_size,
                              hipStream_t stream) {
    const float* x          = (const float*)d_in[0];   // (16384, 1024)
    const float* perm_logit = (const float*)d_in[1];   // (2, 3)
    const float* abcd       = (const float*)d_in[2];   // (2, 4092, 2)
    const float* bias       = (const float*)d_in[3];   // (1024,)
    float* out              = (float*)d_out;           // (16384, 1024)

    // Workspace: W^T split planes, 2 x 1024x1024 bf16 = 4 MB.
    __bf16* wt_hi = (__bf16*)d_ws;
    __bf16* wt_lo = wt_hi + (size_t)NN * NN;

    build_wt_kernel<<<NN, 256, 0, stream>>>(perm_logit, abcd, wt_hi, wt_lo);

    dim3 grid(BATCH / BM, NN / BN);   // 128 x 16
    gemm_wmma_kernel<<<grid, dim3(256), 0, stream>>>(x, wt_hi, wt_lo, bias, out);
}
